// model_FABind_layer_20418274525609
// MI455X (gfx1250) — matrix operand order repaired and checked by reading, not compiled, not run on hardware
//
#include <hip/hip_runtime.h>
#include <math.h>

typedef __attribute__((ext_vector_type(16))) _Float16 v16h;
typedef __attribute__((ext_vector_type(8)))  float    v8f;

// Problem dims (uniform batching)
constexpr int kB   = 16;
constexpr int kNC  = 64;
constexpr int kNP  = 384;
constexpr int kH   = 128;
constexpr int kCR  = 2 + kNC + kNP;   // 450 rows per graph in complex_coords
constexpr int kCF  = 56;
constexpr int kPF  = 1280;
constexpr float kLnEps = 1e-5f;
constexpr int kIPB = 8;               // pocket residues per block in pair kernel

// d_out layout (tuple concatenated flat, in return order)
constexpr int kNPair    = kB * kNP * kNC;         // 393216
constexpr int kOffBatch = 3072;                   // batch ids (as float values)
constexpr int kOffPair  = 4096;
constexpr int kOffDist  = kOffPair + kNPair;      // 397312
constexpr int kOffDmap  = kOffDist + kNPair;      // 790528

constexpr int kW1Elems = kH * kH;                 // 16384
constexpr int kWpElems = kPF * kH;                // 163840

// ---------------------------------------------------------------------------
// Kernel 0: pre-swizzle W1 and Wp (f32 row-major) into the WMMA B-fragment
// f16 layout: idx = ((kk*8+t)*32 + lane)*16 + h ; K = kk*32 + 8*(lane>=16) + (h<8 ? h : h+8),
// N = t*16 + (lane&15). One-time cost; both buffers stay L2-resident.
// ---------------------------------------------------------------------------
__global__ void preswizzle_kernel(const float* __restrict__ W1,
                                  const float* __restrict__ Wp,
                                  _Float16* __restrict__ W1sw,
                                  _Float16* __restrict__ Wpsw) {
  int gid = blockIdx.x * blockDim.x + threadIdx.x;
  const float* src;
  _Float16* dst;
  int idx;
  if (gid < kW1Elems) { src = W1; dst = W1sw; idx = gid; }
  else {
    idx = gid - kW1Elems;
    if (idx >= kWpElems) return;
    src = Wp; dst = Wpsw;
  }
  const int h = idx & 15, ln2 = (idx >> 4) & 31, frag = idx >> 9;
  const int t = frag & 7, kk = frag >> 3;
  const int K = kk * 32 + ((ln2 >= 16) ? 8 : 0) + ((h < 8) ? h : h + 8);
  const int N = t * 16 + (ln2 & 15);
  dst[idx] = (_Float16)src[K * kH + N];
}

// ---------------------------------------------------------------------------
// Kernel 1: compound linear (K=56) + LayerNorm -> c_ln (f32)  [1024 x 128]
// ---------------------------------------------------------------------------
__global__ void compound_ln_kernel(const float* __restrict__ feats,
                                   const float* __restrict__ Wc,
                                   const float* __restrict__ bc,
                                   const float* __restrict__ g,
                                   const float* __restrict__ bln,
                                   float* __restrict__ c_ln) {
  const int row = blockIdx.x;        // 0..B*NC-1
  const int n   = threadIdx.x;       // 0..127
  const float* f = feats + row * kCF;
  float e = bc[n];
#pragma unroll
  for (int k = 0; k < kCF; ++k) e = fmaf(f[k], Wc[k * kH + n], e);

  __shared__ float red[kH];
  red[n] = e; __syncthreads();
  for (int s = 64; s > 0; s >>= 1) { if (n < s) red[n] += red[n + s]; __syncthreads(); }
  const float mean = red[0] * (1.0f / kH);
  __syncthreads();
  const float d = e - mean;
  red[n] = d * d; __syncthreads();
  for (int s = 64; s > 0; s >>= 1) { if (n < s) red[n] += red[n + s]; __syncthreads(); }
  const float rstd = rsqrtf(red[0] * (1.0f / kH) + kLnEps);
  c_ln[row * kH + n] = d * rstd * g[n] + bln[n];
}

// ---------------------------------------------------------------------------
// Kernel 2: pocket linear [6144,1280]x[1280,128] via WMMA + LayerNorm -> p_ln
// B fragments read directly from pre-swizzled Wp (L2-resident), software-
// pipelined: all 8 fragments of a K-chunk loaded into distinct registers
// before the 8 WMMAs, so loads overlap matrix-op execution.
// ---------------------------------------------------------------------------
__global__ void pocket_ln_kernel(const float* __restrict__ feats,
                                 const _Float16* __restrict__ Wpsw,
                                 const float* __restrict__ bp,
                                 const float* __restrict__ g,
                                 const float* __restrict__ bln,
                                 float* __restrict__ p_ln) {
  const int tid  = threadIdx.x;
  const int lane = tid & 31;
  const int wid  = tid >> 5;
  const int rowBase = (blockIdx.x * 4 + wid) * 16;
  const float* af = feats + (size_t)(rowBase + (lane & 15)) * kPF;

  v8f zero = {};
  v8f acc[8];
#pragma unroll
  for (int t = 0; t < 8; ++t) acc[t] = zero;

  for (int kk = 0; kk < kPF / 32; ++kk) {
    const int K0 = kk * 32 + ((lane >> 4) << 3);
    v16h a;
#pragma unroll
    for (int u = 0; u < 8; ++u) a[u]     = (_Float16)af[K0 + u];
#pragma unroll
    for (int u = 0; u < 8; ++u) a[8 + u] = (_Float16)af[K0 + 16 + u];

    const _Float16* bbase = Wpsw + ((size_t)(kk * 8) * 32 + lane) * 16;
    v16h bf[8];                         // independent regs -> loads pipeline
#pragma unroll
    for (int t = 0; t < 8; ++t) bf[t] = *(const v16h*)(bbase + t * 512);
#pragma unroll
    for (int t = 0; t < 8; ++t)
      acc[t] = __builtin_amdgcn_wmma_f32_16x16x32_f16(false, a, false, bf[t],
                                                      (short)0, acc[t], false, false);
  }

  // bias + LayerNorm in registers (row M = r + 8*(lane>=16), col = 16t + lane%16)
  float bpv[8], gv[8], bv[8];
#pragma unroll
  for (int t = 0; t < 8; ++t) {
    const int n = t * 16 + (lane & 15);
    bpv[t] = bp[n]; gv[t] = g[n]; bv[t] = bln[n];
  }
  float x[8][8];
#pragma unroll
  for (int t = 0; t < 8; ++t)
#pragma unroll
    for (int r = 0; r < 8; ++r) x[t][r] = acc[t][r] + bpv[t];

  const int halfOff = (lane >> 4) << 3;
#pragma unroll
  for (int r = 0; r < 8; ++r) {
    float s = 0.f;
#pragma unroll
    for (int t = 0; t < 8; ++t) s += x[t][r];
    s += __shfl_xor(s, 8, 16); s += __shfl_xor(s, 4, 16);
    s += __shfl_xor(s, 2, 16); s += __shfl_xor(s, 1, 16);
    const float mean = s * (1.0f / kH);
    float q = 0.f;
#pragma unroll
    for (int t = 0; t < 8; ++t) { const float d = x[t][r] - mean; q = fmaf(d, d, q); }
    q += __shfl_xor(q, 8, 16); q += __shfl_xor(q, 4, 16);
    q += __shfl_xor(q, 2, 16); q += __shfl_xor(q, 1, 16);
    const float rstd = rsqrtf(q * (1.0f / kH) + kLnEps);
    float* op = p_ln + (size_t)(rowBase + r + halfOff) * kH;
#pragma unroll
    for (int t = 0; t < 8; ++t)
      op[t * 16 + (lane & 15)] = (x[t][r] - mean) * rstd * gv[t] + bv[t];
  }
}

// ---------------------------------------------------------------------------
// Kernel 3 (main): fused pair outer-product MLP.
// h = relu((p_ln ⊙ c_ln) @ W1 + b1); logit = h @ W2 + b2; out = 10*sigmoid.
// One block handles (b, 8 pocket residues); 4 waves x 16 compound atoms each.
// W1 (pre-swizzled, 32KB) copied to LDS via async-to-LDS; c_ln tile held in
// f16 registers across the i-loop; A fragments via v_pk_mul_f16; B fragments
// register-staged per K-chunk so ds_loads overlap WMMA execution.
// ---------------------------------------------------------------------------
__global__ void pair_mlp_kernel(const float* __restrict__ c_ln,
                                const float* __restrict__ p_ln,
                                const _Float16* __restrict__ W1sw,
                                const float* __restrict__ b1,
                                const float* __restrict__ W2,
                                const float* __restrict__ b2,
                                float* __restrict__ out_pair) {
  __shared__ __align__(32) _Float16 sW1[32 * 32 * 16];  // 32KB, B-frag layout
  const int tid  = threadIdx.x;
  const int lane = tid & 31;
  const int wid  = tid >> 5;
  const int bb   = blockIdx.x / (kNP / kIPB);                 // batch graph
  const int i0   = (blockIdx.x % (kNP / kIPB)) * kIPB;        // first pocket idx

  // Async copy of pre-swizzled W1 (pure 32KB copy, 16B x 16 per thread).
  {
    uint32_t ldsaddr = (uint32_t)(uintptr_t)(&sW1[0]) + (uint32_t)tid * 16u;
    uint64_t gaddr   = (uint64_t)(uintptr_t)W1sw + (uint64_t)tid * 16u;
#pragma unroll
    for (int it = 0; it < 16; ++it) {
      asm volatile("global_load_async_to_lds_b128 %0, %1, off"
                   :: "v"(ldsaddr), "v"(gaddr) : "memory");
      ldsaddr += 2048u;
      gaddr   += 2048u;
    }
    asm volatile("s_wait_asynccnt 0" ::: "memory");
  }
  __syncthreads();

  // Hoist this wave's 16 compound rows of c_ln as f16 A-side fragments.
  const int j0 = wid * 16;
  const float* crow = c_ln + (size_t)(bb * kNC + j0 + (lane & 15)) * kH;
  v16h cf[4];
#pragma unroll
  for (int kk = 0; kk < 4; ++kk) {
    const int K0 = kk * 32 + ((lane >> 4) << 3);
#pragma unroll
    for (int u = 0; u < 8; ++u) cf[kk][u]     = (_Float16)crow[K0 + u];
#pragma unroll
    for (int u = 0; u < 8; ++u) cf[kk][8 + u] = (_Float16)crow[K0 + 16 + u];
  }

  float b1v[8], w2v[8];
#pragma unroll
  for (int t = 0; t < 8; ++t) {
    const int n = t * 16 + (lane & 15);
    b1v[t] = b1[n]; w2v[t] = W2[n];
  }
  const float b2v = b2[0];
  const int halfOff = (lane >> 4) << 3;

  for (int ii = 0; ii < kIPB; ++ii) {
    const int bi = bb * kNP + i0 + ii;
    const float* prow = p_ln + (size_t)bi * kH;

    v8f zero = {};
    v8f acc[8];
#pragma unroll
    for (int t = 0; t < 8; ++t) acc[t] = zero;

#pragma unroll
    for (int kk = 0; kk < 4; ++kk) {
      const int K0 = kk * 32 + ((lane >> 4) << 3);
      v16h pf;   // broadcast loads (1 distinct address per wave half)
#pragma unroll
      for (int u = 0; u < 8; ++u) pf[u]     = (_Float16)prow[K0 + u];
#pragma unroll
      for (int u = 0; u < 8; ++u) pf[8 + u] = (_Float16)prow[K0 + 16 + u];
      const v16h a = cf[kk] * pf;            // v_pk_mul_f16: z-row fragment

      v16h bf[8];                            // register-staged ds_loads
#pragma unroll
      for (int t = 0; t < 8; ++t)
        bf[t] = *(const v16h*)(&sW1[((kk * 8 + t) * 32 + lane) * 16]);
#pragma unroll
      for (int t = 0; t < 8; ++t)
        acc[t] = __builtin_amdgcn_wmma_f32_16x16x32_f16(false, a, false, bf[t],
                                                        (short)0, acc[t], false, false);
    }

    // fused epilogue: +b1, relu, dot with W2, 16-lane reduce, sigmoid*10
    float part[8];
#pragma unroll
    for (int r = 0; r < 8; ++r) part[r] = 0.f;
#pragma unroll
    for (int t = 0; t < 8; ++t)
#pragma unroll
      for (int r = 0; r < 8; ++r) {
        float x = acc[t][r] + b1v[t];
        x = fmaxf(x, 0.f);
        part[r] = fmaf(x, w2v[t], part[r]);
      }
#pragma unroll
    for (int r = 0; r < 8; ++r) {
      part[r] += __shfl_xor(part[r], 8, 16);
      part[r] += __shfl_xor(part[r], 4, 16);
      part[r] += __shfl_xor(part[r], 2, 16);
      part[r] += __shfl_xor(part[r], 1, 16);
    }
    if ((lane & 15) == 0) {
      float* op = out_pair + ((size_t)bi << 6) + j0 + halfOff;  // 8 contiguous
#pragma unroll
      for (int r = 0; r < 8; ++r) {
        const float logit = part[r] + b2v;
        op[r] = 10.f / (1.f + __expf(-logit));
      }
    }
  }
}

// ---------------------------------------------------------------------------
// Kernel 4: elementwise outputs (coords passthrough, batch ids, clipped
// true distances, dis_map copy). SCALE cancels: coords_out*5 == raw coords.
// ---------------------------------------------------------------------------
__global__ void elementwise_kernel(const float* __restrict__ complex_coords,
                                   const float* __restrict__ pocket_xyz,
                                   const float* __restrict__ dis_map,
                                   float* __restrict__ out) {
  const int gid = blockIdx.x * blockDim.x + threadIdx.x;
  if (gid >= kNPair) return;
  const int b   = gid / (kNP * kNC);
  const int rem = gid - b * (kNP * kNC);
  const int ip  = rem >> 6;   // / NC
  const int jc  = rem & 63;   // % NC
  const float* p = pocket_xyz + (size_t)(b * kNP + ip) * 3;
  const float* c = complex_coords + (size_t)(b * kCR + 1 + jc) * 3;
  const float dx = p[0] - c[0], dy = p[1] - c[1], dz = p[2] - c[2];
  const float d = sqrtf(dx * dx + dy * dy + dz * dz);
  out[kOffDist + gid] = fminf(d, 10.f);          // clip(pc_dis*SCALE, 0, 10)
  out[kOffDmap + gid] = dis_map[gid];            // passthrough
  if (gid < kOffBatch) {                         // compound_coord_pred
    const int bbx = gid / (kNC * 3);
    const int r2 = gid - bbx * (kNC * 3);
    const int j = r2 / 3, cc = r2 - j * 3;
    out[gid] = complex_coords[(bbx * kCR + 1 + j) * 3 + cc];
  }
  if (gid < kB * kNC)                            // compound_batch ids
    out[kOffBatch + gid] = (float)(gid >> 6);
}

// ---------------------------------------------------------------------------
extern "C" void kernel_launch(void* const* d_in, const int* in_sizes, int n_in,
                              void* d_out, int out_size, void* d_ws, size_t ws_size,
                              hipStream_t stream) {
  const float* compound_feats = (const float*)d_in[0];
  const float* pocket_feats   = (const float*)d_in[1];
  const float* complex_coords = (const float*)d_in[2];
  const float* pocket_xyz     = (const float*)d_in[3];
  const float* dis_map        = (const float*)d_in[4];
  const float* Wc   = (const float*)d_in[5];
  const float* bc   = (const float*)d_in[6];
  const float* Wp   = (const float*)d_in[7];
  const float* bp   = (const float*)d_in[8];
  const float* ln_g = (const float*)d_in[9];
  const float* ln_b = (const float*)d_in[10];
  const float* W1   = (const float*)d_in[11];
  const float* b1   = (const float*)d_in[12];
  const float* W2   = (const float*)d_in[13];
  const float* b2   = (const float*)d_in[14];
  float* out = (float*)d_out;

  float* c_ln = (float*)d_ws;                        // [1024,128] f32 = 512KB
  float* p_ln = c_ln + (size_t)kB * kNC * kH;        // [6144,128] f32 = 3MB
  _Float16* W1sw = (_Float16*)(p_ln + (size_t)kB * kNP * kH);  // 32KB
  _Float16* Wpsw = W1sw + kW1Elems;                             // 320KB

  preswizzle_kernel<<<dim3((kW1Elems + kWpElems + 255) / 256), dim3(256), 0, stream>>>(
      W1, Wp, W1sw, Wpsw);
  compound_ln_kernel<<<dim3(kB * kNC), dim3(kH), 0, stream>>>(
      compound_feats, Wc, bc, ln_g, ln_b, c_ln);
  pocket_ln_kernel<<<dim3(kB * kNP / 64), dim3(128), 0, stream>>>(
      pocket_feats, Wpsw, bp, ln_g, ln_b, p_ln);
  pair_mlp_kernel<<<dim3(kB * (kNP / kIPB)), dim3(128), 0, stream>>>(
      c_ln, p_ln, W1sw, b1, W2, b2, out + kOffPair);
  elementwise_kernel<<<dim3((kNPair + 255) / 256), dim3(256), 0, stream>>>(
      complex_coords, pocket_xyz, dis_map, out);
}
